// DSB_10797547782413
// MI455X (gfx1250) — compile-verified, matrix operand order checked by reading, not run on hardware
//
#include <hip/hip_runtime.h>
#include <hip/hip_bf16.h>
#include <math.h>
#include <stdint.h>

// ---------------- problem constants (from reference) ----------------
#define DIMC   256
#define HEADS  8
#define BB     4
#define HH     64
#define WW     64
#define HWSZ   (HH * WW)      // 4096
#define HIDC   1024
#define EPSV   1e-5f

// ---------------- vector types ----------------
typedef __attribute__((ext_vector_type(16))) _Float16 v16h;
typedef __attribute__((ext_vector_type(8)))  float    v8f;
typedef __attribute__((ext_vector_type(4)))  unsigned u32x4;
typedef __attribute__((ext_vector_type(8)))  int      i32x8;
typedef __attribute__((ext_vector_type(4)))  int      i32x4;

// ---------------- TDM availability (per compile pass) ----------------
#if defined(__has_builtin)
#  if __has_builtin(__builtin_amdgcn_tensor_load_to_lds) && \
      __has_builtin(__builtin_amdgcn_s_wait_tensorcnt)
#    define HAVE_TDM 1
#  endif
#endif
#ifndef HAVE_TDM
#  define HAVE_TDM 0
#endif

#if HAVE_TDM
// Pack a 2D tile descriptor (D#) per CDNA5 ISA ch.8 and issue TENSOR_LOAD_TO_LDS.
// data_size = 2 bytes (f16). Hardware zero-fills reads beyond tensor_dim -> free
// M-edge handling (e.g. the 18-row off2 weight matrix).
// NOTE: this toolchain exposes the 6-arg builtin:
//   (u32x4 group0, i32x8 group1, i32x4 group2, i32x4 group3, i32x8 extra, i32 cpol)
__device__ inline void tdm_load_2d_f16(unsigned lds_off, const void* gaddr,
                                       unsigned tensor_d0, unsigned tensor_d1,
                                       unsigned tile_d0, unsigned tile_d1,
                                       unsigned stride_elems) {
  unsigned long long ga = (unsigned long long)gaddr;
  u32x4 g0;
  g0[0] = 1u;                                   // count=1, user descriptor
  g0[1] = lds_off;                              // lds_addr (bytes)
  g0[2] = (unsigned)(ga & 0xFFFFFFFFu);         // global_addr[31:0]
  g0[3] = (unsigned)((ga >> 32) & 0x01FFFFFFu)  // global_addr[56:32]
        | (2u << 30);                           // type = 2 ("image")
  i32x8 g1;
  g1[0] = (int)(1u << 16);                                      // data_size=1 -> 2B
  g1[1] = (int)((tensor_d0 & 0xFFFFu) << 16);                   // dim0[15:0] @ b48
  g1[2] = (int)(((tensor_d0 >> 16) & 0xFFFFu) |
                ((tensor_d1 & 0xFFFFu) << 16));                 // dim0 hi / dim1 lo
  g1[3] = (int)(((tensor_d1 >> 16) & 0xFFFFu) |
                ((tile_d0 & 0xFFFFu) << 16));                   // dim1 hi / tile0
  g1[4] = (int)(tile_d1 & 0xFFFFu);                             // tile1, tile2=0
  g1[5] = (int)stride_elems;                                    // dim0 stride lo32
  g1[6] = 0;                                                    // stride hi / dim1 stride
  g1[7] = 0;
  i32x4 z4; z4[0] = 0; z4[1] = 0; z4[2] = 0; z4[3] = 0;
  i32x8 z8;
  z8[0] = 0; z8[1] = 0; z8[2] = 0; z8[3] = 0;
  z8[4] = 0; z8[5] = 0; z8[6] = 0; z8[7] = 0;
  __builtin_amdgcn_tensor_load_to_lds(g0, g1, z4, z4, z8, 0);
}
#endif

// =====================================================================
// Generic WMMA GEMM:  out[b] = act( (A(MxK) * B_b(KxN) [+prev] [+bias]) * scale ) [+resid]
//   A: f16 row-major (weights, shared across batch)
//   B: f16 row-major KxN per batch (activations, channel-major NCHW -> K=C, N=HW)
//   nshift: column shift of B reads (implements the 3x3 conv taps)
// Tile: BM=64, BN=128, BK=32; 256 threads = 8 wave32 waves (2x4), 2x2 frags/wave.
// =====================================================================
__launch_bounds__(256)
__global__ void wmma_gemm(const _Float16* __restrict__ A,
                          const _Float16* __restrict__ Bm, long long bsB,
                          const float* __restrict__ bias,
                          float* accf, long long bsAcc, int accum,
                          _Float16* outh, long long bsOuth,
                          const float* __restrict__ resid, long long bsRes,
                          int M, int N, int K, int nshift,
                          float scale, int act) {
  constexpr int BM = 64, BN = 128, BK = 32;
  __shared__ __align__(32) _Float16 As[BM * BK];
  __shared__ __align__(32) _Float16 Bs[BN * BK];   // stored transposed: [n][k]

  const int tid  = threadIdx.x;
  const int lane = tid & 31;
  const int wave = tid >> 5;
  const int wm   = (wave & 1) * 32;     // wave row offset in tile
  const int wn   = (wave >> 1) * 32;    // wave col offset in tile
  const int m0   = blockIdx.y * BM;
  const int n0   = blockIdx.x * BN;
  const int b    = blockIdx.z;

  const _Float16* Bb = Bm + (long long)b * bsB;

  v8f acc00 = {}, acc01 = {}, acc10 = {}, acc11 = {};

  for (int k0 = 0; k0 < K; k0 += BK) {
    // ---- stage A tile (BM x BK) via Tensor Data Mover (wave 0 issues) ----
#if HAVE_TDM
    if (wave == 0) {
      unsigned ldsoff = (unsigned)(unsigned long long)(uintptr_t)(&As[0]);
      const void* g = (const void*)(A + (long long)m0 * K + k0);
      int remM = M - m0; if (remM < 0) remM = 0;
      tdm_load_2d_f16(ldsoff, g, (unsigned)(K - k0), (unsigned)remM,
                      (unsigned)BK, (unsigned)BM, (unsigned)K);
    }
#else
    for (int t = tid; t < BM * BK; t += 256) {
      int r = t / BK, c = t % BK;
      int gr = m0 + r;
      As[t] = (gr < M) ? A[(long long)gr * K + (k0 + c)] : (_Float16)0.f;
    }
#endif
    // ---- stage B tile transposed: Bs[n][k] = B[(k0+k)][n0+n+shift] ----
    for (int t = tid; t < BN * BK; t += 256) {
      int n = t / BK, k = t % BK;
      int gc = n0 + n + nshift;
      gc = gc < 0 ? 0 : (gc >= N ? N - 1 : gc);
      Bs[t] = Bb[(long long)(k0 + k) * N + gc];
    }
    if (k0 + BK < K)
      __builtin_prefetch((const void*)(Bb + (long long)(k0 + BK) * N + n0), 0, 1);
#if HAVE_TDM
    __builtin_amdgcn_s_wait_tensorcnt(0);
#endif
    __syncthreads();

    // ---- fragments + 4 WMMAs per wave per K-step ----
    const int hsel = lane >> 4;      // 0: k 0..15, 1: k 16..31
    const int l16  = lane & 15;
    v16h a0 = *(const v16h*)&As[(wm +      l16) * BK + hsel * 16];
    v16h a1 = *(const v16h*)&As[(wm + 16 + l16) * BK + hsel * 16];
    v16h b0 = *(const v16h*)&Bs[(wn +      l16) * BK + hsel * 16];
    v16h b1 = *(const v16h*)&Bs[(wn + 16 + l16) * BK + hsel * 16];
    acc00 = __builtin_amdgcn_wmma_f32_16x16x32_f16(false, a0, false, b0, (short)0, acc00, false, false);
    acc01 = __builtin_amdgcn_wmma_f32_16x16x32_f16(false, a0, false, b1, (short)0, acc01, false, false);
    acc10 = __builtin_amdgcn_wmma_f32_16x16x32_f16(false, a1, false, b0, (short)0, acc10, false, false);
    acc11 = __builtin_amdgcn_wmma_f32_16x16x32_f16(false, a1, false, b1, (short)0, acc11, false, false);
    __syncthreads();
  }

  // ---- epilogue: C layout (VGPR r -> rows r / r+8 for lane halves) ----
  const int l16 = lane & 15;
  const int rb  = (lane >> 4) * 8;
  for (int i = 0; i < 2; ++i) {
    for (int j = 0; j < 2; ++j) {
      v8f a = (i == 0) ? (j == 0 ? acc00 : acc01) : (j == 0 ? acc10 : acc11);
      int rowb = m0 + wm + i * 16 + rb;
      int col  = n0 + wn + j * 16 + l16;
      for (int r = 0; r < 8; ++r) {
        int row = rowb + r;
        if (row >= M) continue;
        long long ofs = (long long)row * N + col;
        float v = a[r];
        if (accum && accf) v += accf[(long long)b * bsAcc + ofs];
        if (bias) v += bias[row];
        v *= scale;
        if (act == 1) v = v > 0.f ? v : 0.f;                               // ReLU
        else if (act == 2) v = 0.5f * v * (1.f + erff(v * 0.70710678f));   // exact GELU
        if (resid) v += resid[(long long)b * bsRes + ofs];
        if (accf) accf[(long long)b * bsAcc + ofs] = v;
        if (outh) outh[(long long)b * bsOuth + ofs] = (_Float16)v;
      }
    }
  }
}

// =====================================================================
// Instance-norm (mean/var over C*H*W per batch): reduce + apply
// =====================================================================
__launch_bounds__(256)
__global__ void gn_reduce(const float* __restrict__ x, float* __restrict__ stats, int perb) {
  int b = blockIdx.x;
  const float* p = x + (long long)b * perb;
  float s = 0.f, s2 = 0.f;
  for (int i = threadIdx.x; i < perb; i += 256) { float v = p[i]; s += v; s2 += v * v; }
  __shared__ float sh[512];
  sh[threadIdx.x] = s; sh[256 + threadIdx.x] = s2;
  __syncthreads();
  for (int st = 128; st > 0; st >>= 1) {
    if (threadIdx.x < st) {
      sh[threadIdx.x] += sh[threadIdx.x + st];
      sh[256 + threadIdx.x] += sh[256 + threadIdx.x + st];
    }
    __syncthreads();
  }
  if (threadIdx.x == 0) { stats[b * 2] = sh[0]; stats[b * 2 + 1] = sh[256]; }
}

__launch_bounds__(256)
__global__ void gn_apply(const float* __restrict__ x, const float* __restrict__ stats,
                         const float* __restrict__ gw, const float* __restrict__ gb,
                         _Float16* __restrict__ out, long long outBStride,
                         int C, int HWn) {
  long long idx = (long long)blockIdx.x * 256 + threadIdx.x;
  long long per = (long long)C * HWn;
  if (idx >= (long long)BB * per) return;
  int b = (int)(idx / per);
  long long r = idx - (long long)b * per;
  int c = (int)(r / HWn);
  float s = stats[b * 2], s2 = stats[b * 2 + 1];
  float mu = s / (float)per;
  float var = s2 / (float)per - mu * mu;
  float v = (x[idx] - mu) * rsqrtf(var + EPSV) * gw[c] + gb[c];
  out[(long long)b * outBStride + r] = (_Float16)v;
}

// =====================================================================
// Elementwise helpers
// =====================================================================
__global__ void f32_to_f16(const float* __restrict__ in, _Float16* __restrict__ out, long long n) {
  long long i = (long long)blockIdx.x * 256 + threadIdx.x;
  if (i < n) out[i] = (_Float16)in[i];
}

// off1_w (256,512,3,3) -> w9 (tap, 256, 512) f16
__global__ void repack_off1(const float* __restrict__ w, _Float16* __restrict__ out) {
  long long i = (long long)blockIdx.x * 256 + threadIdx.x;
  const long long total = 9LL * DIMC * (2 * DIMC);
  if (i >= total) return;
  int tap = (int)(i / (DIMC * 2 * DIMC));
  long long rem = i - (long long)tap * (DIMC * 2 * DIMC);
  int o = (int)(rem / (2 * DIMC));
  int c = (int)(rem - (long long)o * (2 * DIMC));
  out[i] = (_Float16)w[((long long)o * (2 * DIMC) + c) * 9 + tap];
}

// =====================================================================
// Deformable window attention: per (b,y,x) thread.
//   keys: bilinear samples of kv at learned offsets, 3x3-shifted window,
//   softmax over 9, weighted sum of values. Output f16 for proj GEMM.
// =====================================================================
__launch_bounds__(256)
__global__ void attn_kernel(const _Float16* __restrict__ qh,
                            const _Float16* __restrict__ kvh,
                            const float* __restrict__ off,
                            _Float16* __restrict__ outh) {
  int idx = blockIdx.x * 256 + threadIdx.x;
  if (idx >= BB * HWSZ) return;
  int b  = idx / HWSZ;
  int yx = idx - b * HWSZ;
  int y  = yx / WW, x = yx - y * WW;

  const _Float16* qb = qh  + (long long)b * DIMC * HWSZ;
  const _Float16* kb = kvh + (long long)b * 2 * DIMC * HWSZ;
  const _Float16* vb = kb + (long long)DIMC * HWSZ;
  const float*    ob = off + (long long)b * 18 * HWSZ;
  const int CH = DIMC / HEADS;   // 32

  for (int h = 0; h < HEADS; ++h) {
    float lg[9]; int valid[9]; int ci[9][4]; float cw[9][4];
    for (int o = 0; o < 9; ++o) {
      int sy = y + o / 3 - 1, sx = x + o % 3 - 1;
      lg[o] = 0.f; valid[o] = 0;
      ci[o][0] = ci[o][1] = ci[o][2] = ci[o][3] = 0;
      cw[o][0] = cw[o][1] = cw[o][2] = cw[o][3] = 0.f;
      if (sy >= 0 && sy < HH && sx >= 0 && sx < WW) {
        valid[o] = 1;
        float px = ob[(long long)(2 * o)     * HWSZ + sy * WW + sx];
        float py = ob[(long long)(2 * o + 1) * HWSZ + sy * WW + sx];
        float y0f = floorf(py), x0f = floorf(px);
        float wy = py - y0f, wx = px - x0f;
        int y0 = (int)y0f, x0 = (int)x0f;
        int y1 = y0 + 1, x1 = x0 + 1;
        int y0c = y0 < 0 ? 0 : (y0 > HH - 1 ? HH - 1 : y0);
        int y1c = y1 < 0 ? 0 : (y1 > HH - 1 ? HH - 1 : y1);
        int x0c = x0 < 0 ? 0 : (x0 > WW - 1 ? WW - 1 : x0);
        int x1c = x1 < 0 ? 0 : (x1 > WW - 1 ? WW - 1 : x1);
        bool iy0 = (y0 >= 0 && y0 < HH), iy1 = (y1 >= 0 && y1 < HH);
        bool ix0 = (x0 >= 0 && x0 < WW), ix1 = (x1 >= 0 && x1 < WW);
        cw[o][0] = (iy0 && ix0) ? (1.f - wy) * (1.f - wx) : 0.f;
        cw[o][1] = (iy0 && ix1) ? (1.f - wy) * wx         : 0.f;
        cw[o][2] = (iy1 && ix0) ? wy * (1.f - wx)         : 0.f;
        cw[o][3] = (iy1 && ix1) ? wy * wx                 : 0.f;
        ci[o][0] = y0c * WW + x0c; ci[o][1] = y0c * WW + x1c;
        ci[o][2] = y1c * WW + x0c; ci[o][3] = y1c * WW + x1c;
        float dot = 0.f;
        for (int c = 0; c < CH; ++c) {
          int ch = c * HEADS + h;
          const _Float16* kp = kb + (long long)ch * HWSZ;
          float kvv = cw[o][0] * (float)kp[ci[o][0]] + cw[o][1] * (float)kp[ci[o][1]]
                    + cw[o][2] * (float)kp[ci[o][2]] + cw[o][3] * (float)kp[ci[o][3]];
          dot += (float)qb[(long long)ch * HWSZ + yx] * kvv;
        }
        lg[o] = dot;
      }
    }
    float mx = lg[0];
    for (int o = 1; o < 9; ++o) mx = lg[o] > mx ? lg[o] : mx;
    float wsm[9], se = 0.f;
    for (int o = 0; o < 9; ++o) { wsm[o] = expf(lg[o] - mx); se += wsm[o]; }
    float inv = 1.f / se;
    float outacc[DIMC / HEADS];
    for (int c = 0; c < CH; ++c) outacc[c] = 0.f;
    for (int o = 0; o < 9; ++o) {
      if (!valid[o]) continue;
      float wgt = wsm[o] * inv;
      for (int c = 0; c < CH; ++c) {
        int ch = c * HEADS + h;
        const _Float16* vp = vb + (long long)ch * HWSZ;
        float vv = cw[o][0] * (float)vp[ci[o][0]] + cw[o][1] * (float)vp[ci[o][1]]
                 + cw[o][2] * (float)vp[ci[o][2]] + cw[o][3] * (float)vp[ci[o][3]];
        outacc[c] += wgt * vv;
      }
    }
    for (int c = 0; c < CH; ++c)
      outh[(long long)b * DIMC * HWSZ + (long long)(c * HEADS + h) * HWSZ + yx]
          = (_Float16)outacc[c];
  }
}

// =====================================================================
// Host: orchestrate the full block pipeline on `stream`
// =====================================================================
extern "C" void kernel_launch(void* const* d_in, const int* in_sizes, int n_in,
                              void* d_out, int out_size, void* d_ws, size_t ws_size,
                              hipStream_t stream) {
  (void)in_sizes; (void)n_in; (void)out_size; (void)ws_size;
  const float* x_q     = (const float*)d_in[0];
  const float* x_kv    = (const float*)d_in[1];
  const float* norm1_w = (const float*)d_in[2];
  const float* norm1_b = (const float*)d_in[3];
  const float* q_w     = (const float*)d_in[4];
  const float* q_b     = (const float*)d_in[5];
  const float* kv_w    = (const float*)d_in[6];
  const float* kv_b    = (const float*)d_in[7];
  const float* off1_w  = (const float*)d_in[8];
  const float* off1_b  = (const float*)d_in[9];
  const float* off2_w  = (const float*)d_in[10];
  const float* off2_b  = (const float*)d_in[11];
  const float* proj_w  = (const float*)d_in[12];
  const float* proj_b  = (const float*)d_in[13];
  const float* norm2_w = (const float*)d_in[14];
  const float* norm2_b = (const float*)d_in[15];
  const float* fc1_w   = (const float*)d_in[16];
  const float* fc1_b   = (const float*)d_in[17];
  const float* fc2_w   = (const float*)d_in[18];
  const float* fc2_b   = (const float*)d_in[19];
  float* outp = (float*)d_out;

  // ---- workspace carve-up ----
  char* ws = (char*)d_ws;
  size_t off = 0;
  auto carve = [&](size_t bytes) -> void* {
    off = (off + 255) & ~(size_t)255;
    void* p = ws + off;
    off += bytes;
    return p;
  };
  const long long HW = HWSZ;
  float*    statsQ  = (float*)carve(2 * BB * sizeof(float));
  float*    statsKV = (float*)carve(2 * BB * sizeof(float));
  float*    stats2  = (float*)carve(2 * BB * sizeof(float));
  _Float16* offin_h = (_Float16*)carve((size_t)BB * 2 * DIMC * HW * 2); // [xq_n ; xkv_n]
  _Float16* q_h     = (_Float16*)carve((size_t)BB * DIMC * HW * 2);
  _Float16* kv_h    = (_Float16*)carve((size_t)BB * 2 * DIMC * HW * 2);
  float*    h1_f    = (float*)carve((size_t)BB * DIMC * HW * 4);
  _Float16* h1_h    = (_Float16*)carve((size_t)BB * DIMC * HW * 2);
  float*    off_f   = (float*)carve((size_t)BB * 18 * HW * 4);
  _Float16* attn_h  = (_Float16*)carve((size_t)BB * DIMC * HW * 2);
  float*    xres_f  = (float*)carve((size_t)BB * DIMC * HW * 4);
  _Float16* xn2_h   = (_Float16*)carve((size_t)BB * DIMC * HW * 2);
  _Float16* fc1_hbuf= (_Float16*)carve((size_t)BB * HIDC * HW * 2);
  _Float16* qw_h    = (_Float16*)carve((size_t)DIMC * DIMC * 2);
  _Float16* kvw_h   = (_Float16*)carve((size_t)2 * DIMC * DIMC * 2);
  _Float16* off2w_h = (_Float16*)carve((size_t)18 * DIMC * 2);
  _Float16* projw_h = (_Float16*)carve((size_t)DIMC * DIMC * 2);
  _Float16* fc1w_h  = (_Float16*)carve((size_t)HIDC * DIMC * 2);
  _Float16* fc2w_h  = (_Float16*)carve((size_t)DIMC * HIDC * 2);
  _Float16* w9_h    = (_Float16*)carve((size_t)9 * DIMC * 2 * DIMC * 2);

  auto cvt = [&](const float* in, _Float16* o, long long n) {
    f32_to_f16<<<dim3((unsigned)((n + 255) / 256)), dim3(256), 0, stream>>>(in, o, n);
  };
  cvt(q_w,    qw_h,    (long long)DIMC * DIMC);
  cvt(kv_w,   kvw_h,   (long long)2 * DIMC * DIMC);
  cvt(off2_w, off2w_h, (long long)18 * DIMC);
  cvt(proj_w, projw_h, (long long)DIMC * DIMC);
  cvt(fc1_w,  fc1w_h,  (long long)HIDC * DIMC);
  cvt(fc2_w,  fc2w_h,  (long long)DIMC * HIDC);
  repack_off1<<<dim3((unsigned)((9LL * DIMC * 2 * DIMC + 255) / 256)), dim3(256), 0, stream>>>(off1_w, w9_h);

  // ---- norm1 on x_q / x_kv -> f16, packed as [xq_n ; xkv_n] per batch ----
  gn_reduce<<<dim3(BB), dim3(256), 0, stream>>>(x_q,  statsQ,  DIMC * HWSZ);
  gn_reduce<<<dim3(BB), dim3(256), 0, stream>>>(x_kv, statsKV, DIMC * HWSZ);
  {
    long long tot = (long long)BB * DIMC * HW;
    dim3 g((unsigned)((tot + 255) / 256));
    gn_apply<<<g, dim3(256), 0, stream>>>(x_q,  statsQ,  norm1_w, norm1_b,
                                          offin_h,              2LL * DIMC * HW, DIMC, HWSZ);
    gn_apply<<<g, dim3(256), 0, stream>>>(x_kv, statsKV, norm1_w, norm1_b,
                                          offin_h + DIMC * HW,  2LL * DIMC * HW, DIMC, HWSZ);
  }

  auto gemm = [&](const _Float16* A, const _Float16* Bp, long long bsB,
                  const float* bias, float* accf, long long bsAcc, int accum,
                  _Float16* outh, long long bsOuth,
                  const float* resid, long long bsRes,
                  int M, int N, int K, int shift, float scale, int act) {
    dim3 g((unsigned)(N / 128), (unsigned)((M + 63) / 64), BB);
    wmma_gemm<<<g, dim3(256), 0, stream>>>(A, Bp, bsB, bias, accf, bsAcc, accum,
                                           outh, bsOuth, resid, bsRes,
                                           M, N, K, shift, scale, act);
  };

  const float qscale = 0.17677669529663687f;  // (DIM/HEADS)^-0.5

  // q = (q_w @ xq_n + q_b) * scale
  gemm(qw_h, offin_h, 2LL * DIMC * HW, q_b, nullptr, 0, 0,
       q_h, (long long)DIMC * HW, nullptr, 0, DIMC, HWSZ, DIMC, 0, qscale, 0);
  // kv = kv_w @ xkv_n + kv_b
  gemm(kvw_h, offin_h + DIMC * HW, 2LL * DIMC * HW, kv_b, nullptr, 0, 0,
       kv_h, 2LL * DIMC * HW, nullptr, 0, 2 * DIMC, HWSZ, DIMC, 0, 1.f, 0);

  // off1: 3x3 conv over [xq_n;xkv_n] as 9 shifted GEMMs accumulating in f32
  for (int tap = 0; tap < 9; ++tap) {
    int dy = tap / 3 - 1, dx = tap % 3 - 1;
    int shift = dy * WW + dx;
    bool last = (tap == 8);
    gemm(w9_h + (long long)tap * DIMC * 2 * DIMC, offin_h, 2LL * DIMC * HW,
         last ? off1_b : nullptr,
         h1_f, (long long)DIMC * HW, tap > 0 ? 1 : 0,
         last ? h1_h : nullptr, (long long)DIMC * HW,
         nullptr, 0,
         DIMC, HWSZ, 2 * DIMC, shift, 1.f, last ? 1 : 0 /*ReLU on last*/);
  }

  // off2: 18-ch offsets (f32)
  gemm(off2w_h, h1_h, (long long)DIMC * HW, off2_b,
       off_f, 18LL * HW, 0, nullptr, 0, nullptr, 0,
       18, HWSZ, DIMC, 0, 1.f, 0);

  // deformable window attention
  attn_kernel<<<dim3((BB * HWSZ + 255) / 256), dim3(256), 0, stream>>>(q_h, kv_h, off_f, attn_h);

  // x = x_q + proj(out)
  gemm(projw_h, attn_h, (long long)DIMC * HW, proj_b,
       xres_f, (long long)DIMC * HW, 0, nullptr, 0,
       x_q, (long long)DIMC * HW,
       DIMC, HWSZ, DIMC, 0, 1.f, 0);

  // norm2(x) -> f16
  gn_reduce<<<dim3(BB), dim3(256), 0, stream>>>(xres_f, stats2, DIMC * HWSZ);
  {
    long long tot = (long long)BB * DIMC * HW;
    gn_apply<<<dim3((unsigned)((tot + 255) / 256)), dim3(256), 0, stream>>>(
        xres_f, stats2, norm2_w, norm2_b, xn2_h, (long long)DIMC * HW, DIMC, HWSZ);
  }

  // h = gelu(fc1 @ xn2 + b)
  gemm(fc1w_h, xn2_h, (long long)DIMC * HW, fc1_b,
       nullptr, 0, 0, fc1_hbuf, (long long)HIDC * HW, nullptr, 0,
       HIDC, HWSZ, DIMC, 0, 1.f, 2 /*GELU*/);

  // out = x + fc2 @ h + b  -> d_out (f32)
  gemm(fc2w_h, fc1_hbuf, (long long)HIDC * HW, fc2_b,
       outp, (long long)DIMC * HW, 0, nullptr, 0,
       xres_f, (long long)DIMC * HW,
       DIMC, HWSZ, HIDC, 0, 1.f, 0);
}
